// BiLSTM_CRF_27135603376334
// MI455X (gfx1250) — matrix operand order repaired and checked by reading, not compiled, not run on hardware
//
#include <hip/hip_runtime.h>
#include <hip/hip_bf16.h>

typedef __attribute__((ext_vector_type(16))) __bf16 v16bf;
typedef __attribute__((ext_vector_type(8)))  float  v8f;

constexpr int cB = 128, cT = 512, cE = 100, cEk = 128, cHD = 64, cG = 256, cK = 12;

union F16x16 { v16bf v; uint4 q[2]; };

__device__ __forceinline__ __bf16 f2bf(float f) {
  union { float f; unsigned u; } x; x.f = f;
  unsigned short s = (unsigned short)((x.u + 0x7FFFu + ((x.u >> 16) & 1u)) >> 16);
  union { unsigned short s; __bf16 b; } y; y.s = s;
  return y.b;
}

__device__ __forceinline__ float fast_rcp(float x) {
#if __has_builtin(__builtin_amdgcn_rcpf)
  return __builtin_amdgcn_rcpf(x);     // v_rcp_f32
#else
  return 1.f / x;
#endif
}

__device__ __forceinline__ float fast_tanh(float x) {
#if __has_builtin(__builtin_amdgcn_tanhf)
  return __builtin_amdgcn_tanhf(x);    // v_tanh_f32 (gfx1250 native trans op)
#elif __has_builtin(__builtin_amdgcn_tanh_f32)
  return __builtin_amdgcn_tanh_f32(x);
#else
  return 1.f - 2.f * fast_rcp(__expf(2.f * x) + 1.f);
#endif
}

__device__ __forceinline__ float sigm(float x) { return fast_rcp(1.f + __expf(-x)); }

// A-fragment (16x32 bf16, row major [16][ld]): lane m=lane&15, half=lane>>4
// element e: k = (e/8)*16 + half*8 + (e%8)
__device__ __forceinline__ v16bf load_a(const __bf16* base, int ld, int kb, int lane) {
  int m = lane & 15, half = (lane >> 4) & 1;
  const __bf16* p = base + m * ld + kb + half * 8;
  F16x16 f;
  f.q[0] = *reinterpret_cast<const uint4*>(p);
  f.q[1] = *reinterpret_cast<const uint4*>(p + 16);
  return f.v;
}

// B-fragment (32x16 bf16) from Bt layout [n][ld] (Bt[n][k] = B[k][n]):
// lane n=n0+(lane&15), half=lane>>4; element e: k = (e/8)*16 + half*8 + (e%8)
__device__ __forceinline__ v16bf load_b(const __bf16* bt, int ld, int n0, int kb, int lane) {
  int n = n0 + (lane & 15), half = (lane >> 4) & 1;
  const __bf16* p = bt + n * ld + kb + half * 8;
  F16x16 f;
  f.q[0] = *reinterpret_cast<const uint4*>(p);
  f.q[1] = *reinterpret_cast<const uint4*>(p + 16);
  return f.v;
}

// ---------------------------------------------------------------- converters
__global__ void conv_wih_kernel(const float* wf, const float* wb, __bf16* out) {
  int i = blockIdx.x * blockDim.x + threadIdx.x;        // [2][256][128]
  if (i >= 2 * cG * cEk) return;
  int k = i & 127, n = (i >> 7) & 255, d = i >> 15;
  const float* w = d ? wb : wf;
  out[i] = f2bf(k < cE ? w[n * cE + k] : 0.f);
}

__global__ void conv_whh_kernel(const float* wf, const float* wb, __bf16* out) {
  int i = blockIdx.x * blockDim.x + threadIdx.x;        // [2][256][64]
  if (i >= 2 * cG * cHD) return;
  int k = i & 63, n = (i >> 6) & 255, d = i >> 14;
  const float* w = d ? wb : wf;
  out[i] = f2bf(w[n * cHD + k]);
}

__global__ void conv_wo_kernel(const float* wo, __bf16* out) {
  int i = blockIdx.x * blockDim.x + threadIdx.x;        // [2][16][64]
  if (i >= 2 * 16 * 64) return;
  int k = i & 63, n = (i >> 6) & 15, d = i >> 10;
  out[i] = f2bf(n < cK ? wo[n * 128 + d * 64 + k] : 0.f);
}

// ------------------------------------------------------- fused BiLSTM kernel
// grid (8 batch-tiles, 2 directions), 256 threads (8 waves).
__global__ __launch_bounds__(256) void bilstm_fused_kernel(
    const int* __restrict__ sent, const float* __restrict__ embed,
    const float* __restrict__ bihf, const float* __restrict__ bhhf,
    const float* __restrict__ bihb, const float* __restrict__ bhhb,
    const float* __restrict__ h0, const float* __restrict__ c0,
    const __bf16* __restrict__ wih_bf, const __bf16* __restrict__ whh_bf,
    const __bf16* __restrict__ wo_bf, float* __restrict__ em) {
  const int mb = blockIdx.x;        // batch tile: rows mb*16 .. mb*16+15
  const int dir = blockIdx.y;       // 0 = fwd, 1 = bwd
  const int tid = threadIdx.x;
  const int wave = tid >> 5, lane = tid & 31;
  const int bbase = mb * 16;

  __shared__ __align__(16) __bf16 sEmb[16][cEk];   // token rows, bf16, K padded
  __shared__ __align__(16) __bf16 sH[16][cHD];     // hidden (bf16 for WMMA A)
  __shared__ __align__(16) float  sC[16][cHD];     // cell state f32
  __shared__ __align__(16) float  sG[16][cG];      // pre-activation gates
  __shared__ __align__(16) float  sBias[cG];

  const __bf16* Wih = wih_bf + dir * cG * cEk;
  const __bf16* Whh = whh_bf + dir * cG * cHD;
  const __bf16* Wo  = wo_bf  + dir * 16 * cHD;
  const float* bih = dir ? bihb : bihf;
  const float* bhh = dir ? bhhb : bhhf;

  sBias[tid] = bih[tid] + bhh[tid];
  for (int idx = tid; idx < 16 * cHD; idx += 256) {
    int m = idx >> 6, k = idx & 63;
    sH[m][k] = f2bf(h0[(dir * cB + bbase + m) * cHD + k]);
    sC[m][k] = c0[(dir * cB + bbase + m) * cHD + k];
  }
  __syncthreads();

  // ---- hoist all weight B-fragments into registers (live across all T steps)
  const int n0a = (wave * 2) * 16, n0b = (wave * 2 + 1) * 16;
  v16bf bE0[4], bE1[4], bH0[2], bH1[2];
#pragma unroll
  for (int q = 0; q < 4; ++q) {
    bE0[q] = load_b(Wih, cEk, n0a, q * 32, lane);
    bE1[q] = load_b(Wih, cEk, n0b, q * 32, lane);
  }
#pragma unroll
  for (int q = 0; q < 2; ++q) {
    bH0[q] = load_b(Whh, cHD, n0a, q * 32, lane);
    bH1[q] = load_b(Whh, cHD, n0b, q * 32, lane);
  }
  v16bf bO[2];
  if (wave == 0) {
#pragma unroll
    for (int q = 0; q < 2; ++q) bO[q] = load_b(Wo, cHD, 0, q * 32, lane);
  }
  const float bias0 = sBias[n0a + (lane & 15)];
  const float bias1 = sBias[n0b + (lane & 15)];

  for (int step = 0; step < cT; ++step) {
    const int tt = dir ? (cT - 1 - step) : step;

    // ---- stage 16 embedding rows (gather + f32->bf16, pad E 100->128)
    for (int idx = tid; idx < 16 * cEk; idx += 256) {
      int m = idx >> 7, k = idx & 127;
      int tok = sent[(bbase + m) * cT + tt];
      sEmb[m][k] = f2bf(k < cE ? embed[tok * cE + k] : 0.f);
    }
    if (tid < 16 && step + 1 < cT) {  // prefetch next step's token rows
      int tn = dir ? (cT - 2 - step) : (step + 1);
      int tok = sent[(bbase + tid) * cT + tn];
      __builtin_prefetch(&embed[tok * cE], 0, 1);
    }
    __syncthreads();

    // ---- load this step's A-fragments once (shared by both N-tiles)
    v16bf aE[4], aH[2];
#pragma unroll
    for (int q = 0; q < 4; ++q) aE[q] = load_a(&sEmb[0][0], cEk, q * 32, lane);
#pragma unroll
    for (int q = 0; q < 2; ++q) aH[q] = load_a(&sH[0][0], cHD, q * 32, lane);

    // ---- g = bias + emb@Wih^T + h@Whh^T    (each wave: 2 N-tiles of 16)
    v8f acc0, acc1;
#pragma unroll
    for (int v = 0; v < 8; ++v) { acc0[v] = bias0; acc1[v] = bias1; }
#pragma unroll
    for (int q = 0; q < 4; ++q) {
      acc0 = __builtin_amdgcn_wmma_f32_16x16x32_bf16(false, aE[q], false, bE0[q],
                                                     (short)0, acc0, false, false);
      acc1 = __builtin_amdgcn_wmma_f32_16x16x32_bf16(false, aE[q], false, bE1[q],
                                                     (short)0, acc1, false, false);
    }
#pragma unroll
    for (int q = 0; q < 2; ++q) {
      acc0 = __builtin_amdgcn_wmma_f32_16x16x32_bf16(false, aH[q], false, bH0[q],
                                                     (short)0, acc0, false, false);
      acc1 = __builtin_amdgcn_wmma_f32_16x16x32_bf16(false, aH[q], false, bH1[q],
                                                     (short)0, acc1, false, false);
    }
    {
      const int n_lo = lane & 15, mh = (lane >> 4) << 3;
#pragma unroll
      for (int v = 0; v < 8; ++v) {
        sG[mh + v][n0a + n_lo] = acc0[v];
        sG[mh + v][n0b + n_lo] = acc1[v];
      }
    }
    __syncthreads();

    // ---- gates (PyTorch order i,f,g,o), update c/h
    for (int idx = tid; idx < 16 * cHD; idx += 256) {
      int m = idx >> 6, k = idx & 63;
      float gi = sG[m][k], gf = sG[m][64 + k], gg = sG[m][128 + k], go = sG[m][192 + k];
      float c = sigm(gf) * sC[m][k] + sigm(gi) * fast_tanh(gg);
      float h = sigm(go) * fast_tanh(c);
      sC[m][k] = c;
      sH[m][k] = f2bf(h);
    }
    __syncthreads();

    // ---- emission contribution: h_tile[16x64] @ Wo_dir^T -> [16x16] (12 used)
    if (wave == 0) {
      v8f acc;
#pragma unroll
      for (int v = 0; v < 8; ++v) acc[v] = 0.f;
#pragma unroll
      for (int q = 0; q < 2; ++q) {
        v16bf a = load_a(&sH[0][0], cHD, q * 32, lane);
        acc = __builtin_amdgcn_wmma_f32_16x16x32_bf16(false, a, false, bO[q],
                                                      (short)0, acc, false, false);
      }
      const int n = lane & 15, mh = (lane >> 4) << 3;
#pragma unroll
      for (int v = 0; v < 8; ++v) {
        int bi = bbase + mh + v;
        em[((size_t)(dir * cB + bi) * cT + tt) * 16 + n] = acc[v];
      }
    }
    // next iteration's first barrier orders emission-read-of-sH vs next gate write
  }
}

// ---------------------------------------------------------------- Viterbi
__global__ __launch_bounds__(32) void viterbi_kernel(
    const float* __restrict__ em, const float* __restrict__ b_out,
    const float* __restrict__ trans, float* __restrict__ out_scores,
    float* __restrict__ out_paths) {
  const int b = blockIdx.x, lane = threadIdx.x;
  __shared__ float sTrans[cK][cK];
  __shared__ float sPrev[cK];
  __shared__ unsigned char sBp[cT][cK];

  const float* ef = em + (size_t)b * cT * 16;
  const float* eb = em + (size_t)(cB + b) * cT * 16;
  for (int i = lane; i < cK * cK; i += 32) (&sTrans[0][0])[i] = trans[i];
  float bo = (lane < cK) ? b_out[lane] : 0.f;
  __syncthreads();

  float prev = -3.0e38f;
  if (lane < cK) prev = ef[lane] + eb[lane] + bo;

  for (int t = 1; t < cT; ++t) {
    if (lane < cK) sPrev[lane] = prev;
    __syncthreads();
    if (lane < cK) {
      float best = -3.0e38f; int bi = 0;
#pragma unroll
      for (int i = 0; i < cK; ++i) {
        float c = sPrev[i] + sTrans[i][lane];
        if (c > best) { best = c; bi = i; }
      }
      prev = best + ef[t * 16 + lane] + eb[t * 16 + lane] + bo;
      sBp[t][lane] = (unsigned char)bi;
    }
    __syncthreads();
  }

  if (lane < cK) sPrev[lane] = prev;
  __syncthreads();
  if (lane == 0) {
    float best = sPrev[0]; int cur = 0;
    for (int i = 1; i < cK; ++i)
      if (sPrev[i] > best) { best = sPrev[i]; cur = i; }
    out_scores[b] = best;
    out_paths[(size_t)b * cT + (cT - 1)] = (float)cur;
    for (int t = cT - 1; t >= 1; --t) {
      cur = sBp[t][cur];
      out_paths[(size_t)b * cT + t - 1] = (float)cur;
    }
  }
}

extern "C" void kernel_launch(void* const* d_in, const int* in_sizes, int n_in,
                              void* d_out, int out_size, void* d_ws, size_t ws_size,
                              hipStream_t stream) {
  const int*   sent  = (const int*)d_in[0];
  const float* embed = (const float*)d_in[1];
  const float* Wihf  = (const float*)d_in[2];
  const float* Whhf  = (const float*)d_in[3];
  const float* bihf  = (const float*)d_in[4];
  const float* bhhf  = (const float*)d_in[5];
  const float* Wihb  = (const float*)d_in[6];
  const float* Whhb  = (const float*)d_in[7];
  const float* bihb  = (const float*)d_in[8];
  const float* bhhb  = (const float*)d_in[9];
  const float* Wout  = (const float*)d_in[10];
  const float* bout  = (const float*)d_in[11];
  const float* trans = (const float*)d_in[12];
  const float* h0    = (const float*)d_in[13];
  const float* c0    = (const float*)d_in[14];

  // workspace layout
  char* ws = (char*)d_ws;
  float*  em     = (float*)ws;                                  // [2][B][T][16] f32 (8 MB)
  size_t  off    = (size_t)2 * cB * cT * 16 * sizeof(float);
  __bf16* wih_bf = (__bf16*)(ws + off);                         // [2][256][128]
  off += (size_t)2 * cG * cEk * sizeof(__bf16);
  __bf16* whh_bf = (__bf16*)(ws + off);                         // [2][256][64]
  off += (size_t)2 * cG * cHD * sizeof(__bf16);
  __bf16* wo_bf  = (__bf16*)(ws + off);                         // [2][16][64]

  conv_wih_kernel<<<(2 * cG * cEk + 255) / 256, 256, 0, stream>>>(Wihf, Wihb, wih_bf);
  conv_whh_kernel<<<(2 * cG * cHD + 255) / 256, 256, 0, stream>>>(Whhf, Whhb, whh_bf);
  conv_wo_kernel<<<(2 * 16 * 64 + 255) / 256, 256, 0, stream>>>(Wout, wo_bf);

  dim3 grid(cB / 16, 2);
  bilstm_fused_kernel<<<grid, 256, 0, stream>>>(sent, embed, bihf, bhhf, bihb, bhhb,
                                                h0, c0, wih_bf, whh_bf, wo_bf, em);

  float* o = (float*)d_out;                 // scores [128], then paths [128*512]
  viterbi_kernel<<<cB, 32, 0, stream>>>(em, bout, trans, o, o + cB);
}
